// Attention_21500606284295
// MI455X (gfx1250) — matrix operand order repaired and checked by reading, not compiled, not run on hardware
//
#include <hip/hip_runtime.h>

// ---------------- problem constants ----------------
constexpr int Bc  = 2;
constexpr int Sc  = 2048;
constexpr int DMc = 1024;
constexpr int Hc  = 16;
constexpr int KHc = 4;
constexpr int HDc = 64;          // head dim
constexpr int NREP = Hc / KHc;   // 4
constexpr int ROWS = Bc * Sc;    // 4096
constexpr int QKVC = Hc * HDc + 2 * KHc * HDc; // 1536

typedef __attribute__((ext_vector_type(16))) _Float16 v16h;
typedef __attribute__((ext_vector_type(8)))  _Float16 v8h;
typedef __attribute__((ext_vector_type(8)))  float    v8f;
typedef int v4i_b128 __attribute__((vector_size(16)));   // matches builtin pointee

#define GLOBAL_AS __attribute__((address_space(1)))
#define LDS_AS    __attribute__((address_space(3)))

// gfx1250 async global->LDS copy (ASYNCcnt path), guarded with fallback.
#if defined(__gfx1250__) && __has_builtin(__builtin_amdgcn_global_load_async_to_lds_b128)
#define USE_ASYNC_LDS 1
#else
#define USE_ASYNC_LDS 0
#endif

// copy 16B (8 halves) from global to LDS, per lane
__device__ __forceinline__ void async_copy_b128(const _Float16* g, _Float16* l) {
#if USE_ASYNC_LDS
  __builtin_amdgcn_global_load_async_to_lds_b128(
      (GLOBAL_AS v4i_b128*)g, (LDS_AS v4i_b128*)l, /*offset=*/0, /*cpol=*/0);
#else
  *(v8h*)l = *(const v8h*)g;
#endif
}

// drain this wave's outstanding async copies (must precede barrier signal)
__device__ __forceinline__ void async_join() {
#if USE_ASYNC_LDS
#if __has_builtin(__builtin_amdgcn_s_wait_asynccnt)
  __builtin_amdgcn_s_wait_asynccnt(0);
#else
  asm volatile("s_wait_asynccnt 0x0" ::: "memory");
#endif
#endif
}

// ---------------- WMMA helpers (gfx1250, wave32) ----------------
__device__ __forceinline__ v8f wmma_f16(v16h a, v16h b, v8f c) {
  // D = A(16x32 f16) x B(32x16 f16) + C(16x16 f32)
  return __builtin_amdgcn_wmma_f32_16x16x32_f16(
      /*neg_a=*/false, a, /*neg_b=*/false, b,
      /*c_mod=*/(short)0, c, /*reuse_a=*/false, /*reuse_b=*/false);
}

// A-fragment 16x32 f16 (row-major source, leading dim ld in halves).
// lane<16: row=lane, K=0..15; lane>=16: row=lane-16, K=16..31
__device__ __forceinline__ v16h ld_a_frag(const _Float16* base, int ld) {
  int lane = threadIdx.x & 31;
  int r  = lane & 15;
  int kb = (lane >> 4) << 4;             // 0 or 16
  const _Float16* p = base + r * ld + kb;
  v8h lo = *(const v8h*)p;               // K = kb..kb+7
  v8h hi = *(const v8h*)(p + 8);         // K = kb+8..kb+15
  return __builtin_shufflevector(lo, hi, 0,1,2,3,4,5,6,7,8,9,10,11,12,13,14,15);
}

// B-fragment 32x16 where B[k][n] = Bmem[n][k] (NT form: contiguous in k).
// lane<16: col=lane, K=0..15; lane>=16: col=lane-16, K=16..31
__device__ __forceinline__ v16h ld_b_frag(const _Float16* base, int ld) {
  int lane = threadIdx.x & 31;
  int c  = lane & 15;
  int kb = (lane >> 4) << 4;             // 0 or 16
  const _Float16* p = base + c * ld + kb;
  v8h lo = *(const v8h*)p;
  v8h hi = *(const v8h*)(p + 8);
  return __builtin_shufflevector(lo, hi, 0,1,2,3,4,5,6,7,8,9,10,11,12,13,14,15);
}

// B-fragment 32x16 where B[k][n] = Bmem[k][n] (NN form: strided gather).
__device__ __forceinline__ v16h ld_b_frag_nn(const _Float16* base, int ld) {
  int lane = threadIdx.x & 31;
  int c  = lane & 15;
  int kb = (lane >> 4) << 4;
  v16h b;
#pragma unroll
  for (int i = 0; i < 16; ++i) b[i] = base[(kb + i) * ld + c];
  return b;
}

// ---------------- fp32 -> fp16 convert ----------------
__global__ __launch_bounds__(256) void cvt_f32_f16(const float* __restrict__ s,
                                                   _Float16* __restrict__ d, int n) {
  int i = blockIdx.x * 256 + threadIdx.x;
  if (i < n) d[i] = (_Float16)s[i];
}

// ---------------- GEMM: C(MxN f32) = A(MxK f16) @ B(NxK f16)^T ----------------
// block tile 128x64, 8 waves, each wave 32x32 via 2x2 WMMA accumulators.
__global__ __launch_bounds__(256) void gemm_nt_f16(const _Float16* __restrict__ A,
                                                   const _Float16* __restrict__ Bm,
                                                   float* __restrict__ C,
                                                   int M, int N, int K) {
  constexpr int BM = 128, BN = 64, BK = 32, LDS_LD = BK + 8; // 40 halves = 80B (16B mult)
  __shared__ __align__(16) _Float16 As[BM * LDS_LD];
  __shared__ __align__(16) _Float16 Bs[BN * LDS_LD];

  const int t = threadIdx.x, w = t >> 5, lane = t & 31;
  const int wr = w >> 1, wc = w & 1;                 // 4x2 wave grid
  const int rowBase = blockIdx.y * BM;
  const int colBase = blockIdx.x * BN;

  v8f acc[2][2] = {};

  const int rowA = t >> 1, segA = t & 1;             // A: 16 halves/thread
  const int rowB = t >> 2, segB = t & 3;             // B: 8 halves/thread

  for (int k0 = 0; k0 < K; k0 += BK) {
    { // stage A tile: 128 rows x 32 halves (async global->LDS on gfx1250)
      const _Float16* g = A + (size_t)(rowBase + rowA) * K + k0 + segA * 16;
      async_copy_b128(g,     As + rowA * LDS_LD + segA * 16);
      async_copy_b128(g + 8, As + rowA * LDS_LD + segA * 16 + 8);
    }
    { // stage B tile: 64 rows x 32 halves
      async_copy_b128(Bm + (size_t)(colBase + rowB) * K + k0 + segB * 8,
                      Bs + rowB * LDS_LD + segB * 8);
    }
    if (k0 + BK < K) { // gfx1250 global_prefetch_b8 of next K-tile
      __builtin_prefetch((const void*)(A  + (size_t)(rowBase + rowA) * K + k0 + BK + segA * 16), 0, 0);
      __builtin_prefetch((const void*)(Bm + (size_t)(colBase + rowB) * K + k0 + BK + segB * 8),  0, 0);
    }
    async_join();
    __syncthreads();

    v16h a0 = ld_a_frag(As + (wr * 32)      * LDS_LD, LDS_LD);
    v16h a1 = ld_a_frag(As + (wr * 32 + 16) * LDS_LD, LDS_LD);
    v16h b0 = ld_b_frag(Bs + (wc * 32)      * LDS_LD, LDS_LD);
    v16h b1 = ld_b_frag(Bs + (wc * 32 + 16) * LDS_LD, LDS_LD);
    acc[0][0] = wmma_f16(a0, b0, acc[0][0]);
    acc[0][1] = wmma_f16(a0, b1, acc[0][1]);
    acc[1][0] = wmma_f16(a1, b0, acc[1][0]);
    acc[1][1] = wmma_f16(a1, b1, acc[1][1]);
    __syncthreads();
  }

  // store per C/D layout: VGPR e -> row e (lanes 0..15) / row e+8 (lanes 16..31)
  const int cc = lane & 15;
  const int rb = (lane >> 4) << 3;
#pragma unroll
  for (int i = 0; i < 2; ++i)
#pragma unroll
    for (int j = 0; j < 2; ++j)
#pragma unroll
      for (int e = 0; e < 8; ++e) {
        int r = rowBase + wr * 32 + i * 16 + rb + e;
        C[(size_t)r * N + colBase + wc * 32 + j * 16 + cc] = acc[i][j][e];
      }
}

// ---------------- RoPE + head-major scatter, f32 -> f16 ----------------
// QKV f32 layout (ROWS, 1536): [0:1024]=Q, [1024:1280]=K, [1280:1536]=V
// Qh: (B,H,S,HD)  Kh/Vh: (B,KH,S,HD), rope on Q,K pairs (2i,2i+1) with freq i.
__global__ __launch_bounds__(256) void rope_scatter(const float* __restrict__ QKV,
                                                    const float* __restrict__ cosT,
                                                    const float* __restrict__ sinT,
                                                    _Float16* __restrict__ Qh,
                                                    _Float16* __restrict__ Kh,
                                                    _Float16* __restrict__ Vh) {
  int gid = blockIdx.x * 256 + threadIdx.x;           // one thread per value pair
  constexpr int PAIRS = QKVC / 2;                     // 768
  if (gid >= ROWS * PAIRS) return;
  int row = gid / PAIRS;
  int p   = gid % PAIRS;
  int b = row >> 11, s = row & (Sc - 1);
  int col = p * 2;
  const float* src = QKV + (size_t)row * QKVC + col;
  float x0 = src[0], x1 = src[1];

  if (col < 1280) {
    _Float16* dst;
    int local;
    if (col < 1024) {
      int h = col >> 6; local = col & 63;
      dst = Qh + (((size_t)b * Hc + h) * Sc + s) * HDc + local;
    } else {
      int c2 = col - 1024; int kh = c2 >> 6; local = c2 & 63;
      dst = Kh + (((size_t)b * KHc + kh) * Sc + s) * HDc + local;
    }
    int fi = local >> 1;
    float c = cosT[s * (HDc / 2) + fi], sn = sinT[s * (HDc / 2) + fi];
    dst[0] = (_Float16)(x0 * c - x1 * sn);
    dst[1] = (_Float16)(x0 * sn + x1 * c);
  } else {
    int c2 = col - 1280; int kh = c2 >> 6; int local = c2 & 63;
    _Float16* dst = Vh + (((size_t)b * KHc + kh) * Sc + s) * HDc + local;
    dst[0] = (_Float16)x0;
    dst[1] = (_Float16)x1;
  }
}

// ---------------- causal flash attention ----------------
// grid (S/128, H, B); 8 waves, each owns 16 q rows; K/V streamed in 32-row chunks.
__global__ __launch_bounds__(256) void attn_flash(const _Float16* __restrict__ Q,
                                                  const _Float16* __restrict__ Kx,
                                                  const _Float16* __restrict__ Vx,
                                                  _Float16* __restrict__ O) {
  constexpr int LDK = HDc + 8;   // 72 halves = 144B (16B mult)
  constexpr int LDP = 32 + 8;    // 40 halves = 80B
  __shared__ __align__(16) _Float16 Ks[32 * LDK];
  __shared__ __align__(16) _Float16 Vs[32 * LDK];
  __shared__ __align__(16) _Float16 Ps[8 * 16 * LDP];

  const int t = threadIdx.x, w = t >> 5, lane = t & 31;
  const int qb = blockIdx.x * 128;
  const int h  = blockIdx.y, b = blockIdx.z;
  const int kh = h / NREP;

  const _Float16* Qbase = Q  + (((size_t)b * Hc  + h)  * Sc + (qb + w * 16)) * HDc;
  const _Float16* Kbase = Kx + (((size_t)b * KHc + kh) * Sc) * HDc;
  const _Float16* Vbase = Vx + (((size_t)b * KHc + kh) * Sc) * HDc;

  // Q fragments (16 rows x 64 d) resident in registers for the whole pass
  v16h aQ0 = ld_a_frag(Qbase,      HDc);   // d = 0..31
  v16h aQ1 = ld_a_frag(Qbase + 32, HDc);   // d = 32..63

  v8f o0 = {}, o1 = {}, o2 = {}, o3 = {};
  float m[8], l[8];
#pragma unroll
  for (int e = 0; e < 8; ++e) { m[e] = -1e30f; l[e] = 0.0f; }

  const int rowoff = qb + w * 16 + ((lane >> 4) << 3); // q row of acc element e is rowoff+e
  const int col0   = lane & 15;
  const int qmaxw  = qb + w * 16 + 15;
  _Float16* pw = Ps + w * 16 * LDP;                    // per-wave P scratch

  const int rowKV = t >> 3, segKV = t & 7;             // 32 rows x 8 segs of 8 halves

  for (int kb0 = 0; kb0 < qb + 128; kb0 += 32) {
    { // stage 32x64 K and V chunks via async global->LDS
      async_copy_b128(Kbase + (size_t)(kb0 + rowKV) * HDc + segKV * 8,
                      Ks + rowKV * LDK + segKV * 8);
      async_copy_b128(Vbase + (size_t)(kb0 + rowKV) * HDc + segKV * 8,
                      Vs + rowKV * LDK + segKV * 8);
    }
    async_join();
    __syncthreads();

    if (kb0 <= qmaxw) {  // this wave has at least one unmasked column
      // S = Q K^T : two 16x16 score tiles (k cols 0..15 / 16..31), K-dim = d (2 chunks)
      v8f s0 = {}, s1 = {};
      s0 = wmma_f16(aQ0, ld_b_frag(Ks,                 LDK), s0);
      s0 = wmma_f16(aQ1, ld_b_frag(Ks + 32,            LDK), s0);
      s1 = wmma_f16(aQ0, ld_b_frag(Ks + 16 * LDK,      LDK), s1);
      s1 = wmma_f16(aQ1, ld_b_frag(Ks + 16 * LDK + 32, LDK), s1);

#pragma unroll
      for (int e = 0; e < 8; ++e) {
        int qr = rowoff + e;
        float v0 = (kb0 + col0      <= qr) ? s0[e] * 0.125f : -1e30f;
        float v1 = (kb0 + 16 + col0 <= qr) ? s1[e] * 0.125f : -1e30f;
        // row max over the 16-lane group (row r is a fixed element index)
        float rm = fmaxf(v0, v1);
        rm = fmaxf(rm, __shfl_xor(rm, 1, 16));
        rm = fmaxf(rm, __shfl_xor(rm, 2, 16));
        rm = fmaxf(rm, __shfl_xor(rm, 4, 16));
        rm = fmaxf(rm, __shfl_xor(rm, 8, 16));
        float mn   = fmaxf(m[e], rm);
        float corr = __expf(m[e] - mn);
        float p0   = __expf(v0 - mn);
        float p1   = __expf(v1 - mn);
        float rs   = p0 + p1;
        rs += __shfl_xor(rs, 1, 16);
        rs += __shfl_xor(rs, 2, 16);
        rs += __shfl_xor(rs, 4, 16);
        rs += __shfl_xor(rs, 8, 16);
        l[e] = l[e] * corr + rs;
        m[e] = mn;
        o0[e] *= corr; o1[e] *= corr; o2[e] *= corr; o3[e] *= corr;
        // D-layout -> row-major P in per-wave LDS (for A-fragment reload)
        int pr = ((lane >> 4) << 3) + e;
        pw[pr * LDP + col0]      = (_Float16)p0;
        pw[pr * LDP + 16 + col0] = (_Float16)p1;
      }

      // O += P(16x32) @ V(32x64): 4 WMMAs, V gathered column-wise (NN form)
      v16h aP = ld_a_frag(pw, LDP);
      o0 = wmma_f16(aP, ld_b_frag_nn(Vs +  0, LDK), o0);
      o1 = wmma_f16(aP, ld_b_frag_nn(Vs + 16, LDK), o1);
      o2 = wmma_f16(aP, ld_b_frag_nn(Vs + 32, LDK), o2);
      o3 = wmma_f16(aP, ld_b_frag_nn(Vs + 48, LDK), o3);
    }
    __syncthreads();
  }

  // normalize and store to (B,S,H,HD) f16 for the output projection
#pragma unroll
  for (int e = 0; e < 8; ++e) {
    float inv = 1.0f / l[e];
    int qr = rowoff + e;
    _Float16* dst = O + (((size_t)b * Sc + qr) * Hc + h) * HDc;
    dst[     col0] = (_Float16)(o0[e] * inv);
    dst[16 + col0] = (_Float16)(o1[e] * inv);
    dst[32 + col0] = (_Float16)(o2[e] * inv);
    dst[48 + col0] = (_Float16)(o3[e] * inv);
  }
}

// ---------------- host launcher ----------------
extern "C" void kernel_launch(void* const* d_in, const int* in_sizes, int n_in,
                              void* d_out, int out_size, void* d_ws, size_t ws_size,
                              hipStream_t stream) {
  const float* x    = (const float*)d_in[0];
  const float* wq   = (const float*)d_in[1];
  const float* wk   = (const float*)d_in[2];
  const float* wv   = (const float*)d_in[3];
  const float* wo   = (const float*)d_in[4];
  const float* fcos = (const float*)d_in[5];
  const float* fsin = (const float*)d_in[6];
  float* out = (float*)d_out;

  // workspace carve-up (all offsets 256B-aligned; total ~57 MB, fits MI455X L2)
  char* ws = (char*)d_ws;
  size_t off = 0;
  _Float16* Xh    = (_Float16*)(ws + off); off += (size_t)ROWS * DMc * 2;          // 8 MiB
  _Float16* Wqkvh = (_Float16*)(ws + off); off += (size_t)QKVC * DMc * 2;          // 3 MiB
  _Float16* Woh   = (_Float16*)(ws + off); off += (size_t)DMc * DMc * 2;           // 2 MiB
  float*    QKVf  = (float*)   (ws + off); off += (size_t)ROWS * QKVC * 4;         // 24 MiB
  _Float16* Qh    = (_Float16*)(ws + off); off += (size_t)Bc * Hc  * Sc * HDc * 2; // 8 MiB
  _Float16* Kh    = (_Float16*)(ws + off); off += (size_t)Bc * KHc * Sc * HDc * 2; // 2 MiB
  _Float16* Vh    = (_Float16*)(ws + off); off += (size_t)Bc * KHc * Sc * HDc * 2; // 2 MiB
  _Float16* Oh    = (_Float16*)(ws + off); off += (size_t)ROWS * DMc * 2;          // 8 MiB
  (void)ws_size; (void)in_sizes; (void)n_in; (void)out_size;

  // 1. fp32 -> fp16 staging
  cvt_f32_f16<<<(ROWS * DMc) / 256, 256, 0, stream>>>(x, Xh, ROWS * DMc);
  cvt_f32_f16<<<(Hc * HDc * DMc) / 256, 256, 0, stream>>>(wq, Wqkvh, Hc * HDc * DMc);
  cvt_f32_f16<<<(KHc * HDc * DMc) / 256, 256, 0, stream>>>(wk, Wqkvh + (size_t)Hc * HDc * DMc,
                                                           KHc * HDc * DMc);
  cvt_f32_f16<<<(KHc * HDc * DMc) / 256, 256, 0, stream>>>(wv, Wqkvh + (size_t)(Hc + KHc) * HDc * DMc,
                                                           KHc * HDc * DMc);
  cvt_f32_f16<<<(DMc * DMc) / 256, 256, 0, stream>>>(wo, Woh, DMc * DMc);

  // 2. fused QKV projection: (4096x1536) = Xh @ [wq;wk;wv]^T
  gemm_nt_f16<<<dim3(QKVC / 64, ROWS / 128), 256, 0, stream>>>(Xh, Wqkvh, QKVf,
                                                               ROWS, QKVC, DMc);

  // 3. RoPE + head-major scatter to f16
  rope_scatter<<<(ROWS * (QKVC / 2)) / 256, 256, 0, stream>>>(QKVf, fcos, fsin, Qh, Kh, Vh);

  // 4. causal flash attention (GQA: 4 q-heads share each K/V head)
  attn_flash<<<dim3(Sc / 128, Hc, Bc), 256, 0, stream>>>(Qh, Kh, Vh, Oh);

  // 5. output projection: (4096x1024) = Oh @ wo^T -> fp32
  gemm_nt_f16<<<dim3(DMc / 64, ROWS / 128), 256, 0, stream>>>(Oh, Woh, out,
                                                              ROWS, DMc, DMc);
}
